// EIModel_84731114816215
// MI455X (gfx1250) — matrix operand order repaired and checked by reading, not compiled, not run on hardware
//
#include <hip/hip_runtime.h>
#include <hip/hip_bf16.h>

typedef __attribute__((ext_vector_type(16))) _Float16 v16h;
typedef __attribute__((ext_vector_type(8)))  float    v8f;

#define NPAD    1024        // padded E == padded I
#define NREAL   1000
#define KCHUNKS 32          // 1024 / 32 (K per wmma)
#define NTILES  64          // 1024 / 16 (N per wmma)

// LIF constants (from reference: DT=1e-3, TAU_MEM_INV=100, TAU_SYN_INV=200)
#define DT_TAU_MEM 0.1f     // DT * TAU_MEM_INV
#define SYN_DECAY  0.8f     // 1 - DT * TAU_SYN_INV
#define V_TH_C     1.0f

// ---------------------------------------------------------------------------
// Pack fp32 weights into WMMA-tile-contiguous f16 layout.
// Logical matvec form: out[n] = sum_k z[k] * M[k][n], K=N=1024 (zero padded).
// Packed element order: [n_tile][k_chunk][lane][h], in this file's contiguous-k
// B-fragment layout (16-bit, 32x16): N = lane&15, K = (lane>>4)*16 + h.
// Each lane loads its whole fragment as one contiguous 32B v16h (2x b128).
// ---------------------------------------------------------------------------
__global__ void pack_weights(const float* __restrict__ src,
                             _Float16* __restrict__ dst, int transpose)
{
    int idx = blockIdx.x * blockDim.x + threadIdx.x;   // 0 .. 1024*1024-1
    if (idx >= NPAD * NPAD) return;
    int h     = idx & 15;
    int lane  = (idx >> 4) & 31;
    int kc    = (idx >> 9) & 31;
    int ntile = idx >> 14;
    int K = kc * 32 + (lane >> 4) * 16 + h;
    int N = ntile * 16 + (lane & 15);
    float v = 0.0f;
    if (K < NREAL && N < NREAL)
        v = transpose ? src[N * NREAL + K] : src[K * NREAL + N];
    dst[idx] = (_Float16)v;
}

// ---------------------------------------------------------------------------
// A fragment: 16x32 f16, every row M = same spike chunk z[k0..k0+31].
// Lane L -> M=L&15, half=L>>4; VGPR g holds K pair (contiguous, matches B pack)
//   K = half*16 + 2g + p
// Broadcast over M: fragment depends only on `half` -> 2x ds_load_b128.
// ---------------------------------------------------------------------------
__device__ __forceinline__ v16h build_a(const _Float16* z, int k0, int half)
{
    union { v16h v; unsigned u[8]; } r;
    const unsigned* zp = (const unsigned*)(z + k0);    // k0 multiple of 32
    const int b = half * 8;
#pragma unroll
    for (int g = 0; g < 4; ++g) r.u[g]     = zp[b + g];
#pragma unroll
    for (int g = 0; g < 4; ++g) r.u[4 + g] = zp[b + 4 + g];
    return r.v;
}

__device__ __forceinline__ v16h load_b(const _Float16* __restrict__ wp,
                                       int tile, int kc, int lane)
{
    return *(const v16h*)(wp + ((((tile * KCHUNKS) + kc) * 32 + lane) << 4));
}

__device__ __forceinline__ v8f wmma_f16(v16h a, v16h b, v8f c)
{
    return __builtin_amdgcn_wmma_f32_16x16x32_f16(
        /*neg_a=*/false, a, /*neg_b=*/false, b,
        /*c_mod=*/(short)0, c, /*reuse_a=*/false, /*reuse_b=*/false);
}

// norse lif_step: decay with OLD i, threshold, reset, then integrate input
__device__ __forceinline__ float lif_update(float c, float* varr, float* iarr, int n)
{
    float v = varr[n], i = iarr[n];
    float v_dec = v + DT_TAU_MEM * (i - v);
    float i_dec = i * SYN_DECAY;
    float z = (v_dec > V_TH_C) ? 1.0f : 0.0f;          // heaviside(v_dec - 1)
    varr[n] = (1.0f - z) * v_dec;                      // v_reset = 0
    iarr[n] = i_dec + c;
    return z;
}

// ---------------------------------------------------------------------------
// Persistent single-workgroup kernel: 1024 threads = 32 wave32s on one WGP.
// Each wave owns neuron tiles 2w and 2w+1 (fixed across all steps). Tile loop
// is OUTER (k-loop holds only 2 acc + 2 A + 2 B fragments -> fits the
// ~128-VGPR/wave budget at 8 waves/SIMD, no spills). LIF is applied directly
// on the WMMA accumulators (D row M=0 = acc[0] in lanes 0..15, N = lane);
// only 2 barriers per timestep (publish E spikes, publish I spikes).
// ---------------------------------------------------------------------------
__global__ void __launch_bounds__(1024, 1)
ei_lif_persistent(const float* __restrict__ xin,
                  const _Float16* __restrict__ wrec_p,   // M[k][n] = w_rec[n][k]
                  const _Float16* __restrict__ wie_p,    // M[k][n] = w_ie[k][n]
                  const _Float16* __restrict__ wei_p,    // M[k][n] = w_ei[k][n]
                  float* __restrict__ out, int T)
{
    __shared__ __align__(16) _Float16 ze[2][NPAD];   // double-buffered E spikes
    __shared__ __align__(16) _Float16 zi[NPAD];      // I spikes
    __shared__ float ve[NPAD], ie[NPAD], vi[NPAD], ii[NPAD];

    const int tid  = threadIdx.x;
    const int lane = tid & 31;
    const int wave = tid >> 5;
    const int half = lane >> 4;

    // zero all state (padded lanes stay 0 forever: their weights are 0)
    ze[0][tid] = (_Float16)0.0f; ze[1][tid] = (_Float16)0.0f;
    zi[tid] = (_Float16)0.0f;
    ve[tid] = 0.0f; ie[tid] = 0.0f; vi[tid] = 0.0f; ii[tid] = 0.0f;
    __syncthreads();

    for (int t = 0; t < T; ++t) {
        const int p = t & 1;
        const _Float16* zeOld = ze[p];

        // ---- phase 1: cur_e = ze_old @ w_rec^T - zi_old @ w_ie, per tile ----
        for (int tt = 0; tt < 2; ++tt) {
            const int tile = wave * 2 + tt;
            v8f acc_rec = {}; v8f acc_inh = {};
            for (int kc = 0; kc < KCHUNKS; ++kc) {
                v16h a_ze  = build_a(zeOld, kc * 32, half);
                v16h a_zi  = build_a(zi,    kc * 32, half);
                v16h b_rec = load_b(wrec_p, tile, kc, lane);
                v16h b_ie  = load_b(wie_p,  tile, kc, lane);
                if (kc + 1 < KCHUNKS) {
                    __builtin_prefetch(wrec_p + ((((tile * KCHUNKS) + kc + 1) * 32 + lane) << 4), 0, 1);
                    __builtin_prefetch(wie_p  + ((((tile * KCHUNKS) + kc + 1) * 32 + lane) << 4), 0, 1);
                }
                acc_rec = wmma_f16(a_ze, b_rec, acc_rec);
                acc_inh = wmma_f16(a_zi, b_ie,  acc_inh);
            }
            // E-pool LIF directly on accumulators (owning lanes 0..15)
            if (lane < 16) {
                int n = tile * 16 + lane;
                float x = (n < NREAL) ? xin[(size_t)t * NREAL + n] : 0.0f;
                float z = lif_update(x + acc_rec[0] - acc_inh[0], ve, ie, n);
                ze[p ^ 1][n] = (_Float16)z;
                if (n < NREAL) out[(size_t)t * NREAL + n] = z;
            }
        }
        __syncthreads();   // publish new E spikes to all waves

        // ---- phase 2: xi = ze_new @ w_ei, per tile ----
        const _Float16* zeNew = ze[p ^ 1];
        for (int tt = 0; tt < 2; ++tt) {
            const int tile = wave * 2 + tt;
            v8f acc = {};
            for (int kc = 0; kc < KCHUNKS; ++kc) {
                v16h a = build_a(zeNew, kc * 32, half);
                v16h b = load_b(wei_p, tile, kc, lane);
                acc = wmma_f16(a, b, acc);
            }
            // I-pool LIF directly on accumulators
            if (lane < 16) {
                int n = tile * 16 + lane;
                zi[n] = (_Float16)lif_update(acc[0], vi, ii, n);
            }
        }
        __syncthreads();   // publish new I spikes before next step
    }
}

// ---------------------------------------------------------------------------
// Host-side launch. setup_inputs order:
//   d_in[0]=input_currents [T,1000] f32, d_in[1]=w_in (identity, unused),
//   d_in[2]=w_rec [E,E], d_in[3]=w_ei [E,I], d_in[4]=w_ie [I,E]
// d_ws: 3 packed f16 matrices, 2 MB each (needs 6 MB scratch).
// ---------------------------------------------------------------------------
extern "C" void kernel_launch(void* const* d_in, const int* in_sizes, int n_in,
                              void* d_out, int out_size, void* d_ws, size_t ws_size,
                              hipStream_t stream)
{
    const float* x     = (const float*)d_in[0];
    const float* w_rec = (const float*)d_in[2];
    const float* w_ei  = (const float*)d_in[3];
    const float* w_ie  = (const float*)d_in[4];

    const int T = in_sizes[0] / NREAL;

    const size_t MATB = (size_t)NPAD * NPAD;          // elements per packed matrix
    _Float16* wrec_p = (_Float16*)d_ws;
    _Float16* wie_p  = wrec_p + MATB;
    _Float16* wei_p  = wie_p  + MATB;

    dim3 pg((unsigned)((MATB + 255) / 256)), pb(256);
    pack_weights<<<pg, pb, 0, stream>>>(w_rec, wrec_p, /*transpose=*/1); // w_rec.T matvec
    pack_weights<<<pg, pb, 0, stream>>>(w_ie,  wie_p,  /*transpose=*/0);
    pack_weights<<<pg, pb, 0, stream>>>(w_ei,  wei_p,  /*transpose=*/0);

    ei_lif_persistent<<<1, 1024, 0, stream>>>(x, wrec_p, wie_p, wei_p,
                                              (float*)d_out, T);
}
